// EdgeTransformer_33397665694632
// MI455X (gfx1250) — matrix operand order repaired and checked by reading, not compiled, not run on hardware
//
#include <hip/hip_runtime.h>
#include <hip/hip_bf16.h>

// ---------------------------------------------------------------------------
// EdgeTransformer for MI455X (gfx1250, wave32, WMMA)
// B=64, N=50, D=128, ROWS = B*N*N = 160000.
// GEMMs: f16 WMMA (v_wmma_f32_16x16x32_f16), f32 accumulate; f32 master copy
// of pf kept for residual/LayerNorm precision. B-matrix fragments are staged
// through LDS with double buffering (8KB slices), shared by all 8 waves.
// ---------------------------------------------------------------------------

typedef __attribute__((ext_vector_type(16))) _Float16 v16h;
typedef __attribute__((ext_vector_type(8)))  _Float16 v8h;
typedef __attribute__((ext_vector_type(4)))  _Float16 v4h;
typedef __attribute__((ext_vector_type(8)))  float    v8f;

union U16 { v16h v; v8h h[2]; };

#define NB   50
#define BB   64
#define DD   128
#define QKVW 512              // fused q|k|v1|v2 row width
#define ROWS 160000           // B*N*N, = 16 * 10000, = 128 * 1250

__device__ inline v8f wmma_f16(v16h a, v16h b, v8f c) {
  // D = A(16x32 f16) * B(32x16 f16) + C(16x16 f32)
  return __builtin_amdgcn_wmma_f32_16x16x32_f16(
      /*neg_a=*/false, a, /*neg_b=*/false, b,
      /*c_mod=*/(short)0, c, /*reuse_a=*/false, /*reuse_b=*/false);
}

// ---------------------------------------------------------------------------
// Weight packing: W (f32, row-major K x N) -> fragment-native f16 layout:
// P[((kt*(N/16)+nt)*32 + lane)*16 + t] = W[kt*32 + (lane<16?0:16) + t][nt*16 + lane%16]
// so a wave's B-fragment is one contiguous v16h per lane, and a block's
// per-ktile slice (8 n-tiles) is 256 contiguous v16h (8KB).
// ---------------------------------------------------------------------------
__global__ void pack_w_kernel(const float* __restrict__ W, _Float16* __restrict__ P,
                              int K, int N) {
  int idx = blockIdx.x * 256 + threadIdx.x;
  if (idx >= K * N) return;
  int t    = idx & 15;
  int lane = (idx >> 4) & 31;
  int rest = idx >> 9;                 // kt*(N/16) + nt
  int ntiles = N >> 4;
  int nt = rest % ntiles;
  int kt = rest / ntiles;
  int kidx = kt * 32 + ((lane < 16) ? 0 : 16) + t;
  int nidx = nt * 16 + (lane & 15);
  P[idx] = (_Float16)W[(size_t)kidx * N + nidx];
}

// Pack four 128x128 weights as one virtual 128x512 fragment-native matrix.
__global__ void pack_qkv_kernel(const float* __restrict__ W0, const float* __restrict__ W1,
                                const float* __restrict__ W2, const float* __restrict__ W3,
                                _Float16* __restrict__ P) {
  int idx = blockIdx.x * 256 + threadIdx.x;     // K*N = 128*512
  if (idx >= 128 * 512) return;
  int t    = idx & 15;
  int lane = (idx >> 4) & 31;
  int rest = idx >> 9;                 // kt*32 + nt  (ntiles = 32)
  int nt = rest & 31;
  int kt = rest >> 5;
  int kidx = kt * 32 + ((lane < 16) ? 0 : 16) + t;
  int n = nt * 16 + (lane & 15);
  const float* W = (n < 128) ? W0 : (n < 256) ? W1 : (n < 384) ? W2 : W3;
  P[idx] = (_Float16)W[(size_t)kidx * 128 + (n & 127)];
}

// ---------------------------------------------------------------------------
// WMMA GEMM: out = A(f16,[ROWS,K]) * Bpacked (+bias)(+resid)(relu?)
// Each wave owns a 16-row x 128-col strip: one A fragment feeds 8 WMMAs per
// k-step (32 WMMAs per wave at K=128). B fragments staged through LDS with
// double buffering: per ktile the block copies one contiguous 8KB slice
// (256 x v16h, one per thread), shared by all 8 waves -> 8x less L2 traffic.
// grid = (ROWS/128, N/128), block = 256 (8 waves).
// ---------------------------------------------------------------------------
__global__ void __launch_bounds__(256)
gemm_wmma_kernel(const _Float16* __restrict__ A, const _Float16* __restrict__ Bp,
                 const float* __restrict__ bias, const float* __restrict__ resid,
                 float* __restrict__ outF, _Float16* __restrict__ outH,
                 int K, int N, int relu) {
  __shared__ __align__(32) _Float16 Bs[2][256 * 16];   // 2 x 8KB slices

  const int tid   = threadIdx.x;
  const int lane  = tid & 31;
  const int wave  = tid >> 5;
  const int mtile = blockIdx.x * 8 + wave;
  const int nbase = blockIdx.y * 8;            // base 16-col tile index
  const int ntiles  = N >> 4;
  const int nktiles = K >> 5;
  const int rowA  = mtile * 16 + (lane & 15);
  const _Float16* Arow = A + (size_t)rowA * K;
  const int asel = (lane < 16) ? 0 : 16;

  const v16h* Bg  = (const v16h*)Bp;           // global packed fragments
  v16h*       Bsv = (v16h*)Bs;                 // [2][256]

  v8f acc[8];
  {
    v8f z = {};
#pragma unroll
    for (int t = 0; t < 8; ++t) acc[t] = z;
  }

  // Prologue: stage ktile 0 slice (contiguous 256 v16h).
  Bsv[tid] = Bg[((size_t)0 * ntiles + nbase) * 32 + tid];
  __syncthreads();

  int buf = 0;
  for (int kt = 0; kt < nktiles; ++kt) {
    const int kk = kt * 32;
    // Prefetch-stage next ktile slice into the other buffer.
    if (kt + 1 < nktiles)
      Bsv[(buf ^ 1) * 256 + tid] =
          Bg[((size_t)(kt + 1) * ntiles + nbase) * 32 + tid];
    // Hint the A stream ahead (global_prefetch_b8).
    if (kk + 64 < K) __builtin_prefetch(Arow + kk + 64, 0, 3);

    U16 ua;
    ua.h[0] = *(const v8h*)(Arow + kk + asel);        // K = kk + {0..7 | 16..23}
    ua.h[1] = *(const v8h*)(Arow + kk + 8 + asel);    // K = kk + 8 + {..}

    const v16h* bc = Bsv + buf * 256;
#pragma unroll
    for (int nt = 0; nt < 8; ++nt)
      acc[nt] = wmma_f16(ua.v, bc[nt * 32 + lane], acc[nt]);

    __syncthreads();
    buf ^= 1;
  }

  const int rbase = mtile * 16 + ((lane < 16) ? 0 : 8);
#pragma unroll
  for (int nt = 0; nt < 8; ++nt) {
    int col = (nbase + nt) * 16 + (lane & 15);
    float bs = bias ? bias[col] : 0.f;
#pragma unroll
    for (int r = 0; r < 8; ++r) {
      int row = rbase + r;
      float x = acc[nt][r] + bs;
      size_t o = (size_t)row * N + col;
      if (resid) x += resid[o];
      if (relu)  x = x > 0.f ? x : 0.f;
      if (outF)  outF[o] = x;
      if (outH)  outH[o] = (_Float16)x;
    }
  }
}

// ---------------------------------------------------------------------------
// Triangle attention, one block per (b,i). qkv rows are 512 wide:
// [q(0:128) | k(128:256) | v1(256:384) | v2(384:512)].
// S = k_i * q_i^T via WMMA in LDS, masked softmax over l<50, then
// msg[j,d] = sum_l att[l,j]*v1[l,d]*v2[b,l,j,d].
// ---------------------------------------------------------------------------
__global__ void __launch_bounds__(256)
attn_kernel(const _Float16* __restrict__ qkv, _Float16* __restrict__ msg) {
  __shared__ __align__(32) _Float16 qS[64 * 128];
  __shared__ __align__(32) _Float16 kS[64 * 128];
  __shared__ __align__(32) _Float16 v1S[50 * 128];
  __shared__ __align__(16) float    att[64 * 64];

  const int bi   = blockIdx.x;          // b*N + i
  const int b    = bi / NB;
  const int tid  = threadIdx.x;
  const int lane = tid & 31;
  const int wave = tid >> 5;

  // Stage q/k (zero-padded to 64 rows) and v1 (50 rows) into LDS.
#pragma unroll
  for (int it = 0; it < 4; ++it) {
    int idx = (tid + it * 256) * 8;     // halves; 8 per thread per iter
    int row = idx >> 7;
    int col = idx & 127;
    v8h zero = {};
    v8h qv = zero, kv = zero;
    if (row < NB) {
      size_t g = ((size_t)(bi * NB + row)) * QKVW + col;
      qv = *(const v8h*)(qkv + g);                // q block
      kv = *(const v8h*)(qkv + g + 128);          // k block
      *(v8h*)(v1S + idx) = *(const v8h*)(qkv + g + 256);  // v1 block
    }
    *(v8h*)(qS + idx) = qv;
    *(v8h*)(kS + idx) = kv;
  }
  __syncthreads();

  // Scores: 4x4 tiles of 16x16 over K=128; 2 tiles per wave.
  const float scale = 0.08838834764831845f;    // 1/sqrt(128)
  const int asel = (lane < 16) ? 0 : 16;
  const int bsel = (lane < 16) ? 0 : 16;
#pragma unroll
  for (int s = 0; s < 2; ++s) {
    int tile = wave * 2 + s;
    int lt = tile >> 2, jt = tile & 3;
    v8f acc = {};
    const _Float16* ar = kS + (lt * 16 + (lane & 15)) * DD;
    const _Float16* br = qS + (jt * 16 + (lane & 15)) * DD;
#pragma unroll
    for (int kk = 0; kk < 128; kk += 32) {
      U16 ua;
      ua.h[0] = *(const v8h*)(ar + kk + asel);
      ua.h[1] = *(const v8h*)(ar + kk + 8 + asel);
      v16h bf = *(const v16h*)(br + kk + bsel);   // B^T contiguous from row-major q
      acc = wmma_f16(ua.v, bf, acc);
    }
    int jj = jt * 16 + (lane & 15);
    int lb = lt * 16 + ((lane < 16) ? 0 : 8);
#pragma unroll
    for (int r = 0; r < 8; ++r) att[(lb + r) * 64 + jj] = acc[r] * scale;
  }
  __syncthreads();

  // Softmax over middle node l (< 50) for each column j.
  if (tid < 64) {
    int j = tid;
    float m = -1e30f;
    for (int l = 0; l < NB; ++l) m = fmaxf(m, att[l * 64 + j]);
    float ssum = 0.f;
    for (int l = 0; l < NB; ++l) {
      float e = __expf(att[l * 64 + j] - m);
      att[l * 64 + j] = e;
      ssum += e;
    }
    float inv = 1.0f / ssum;
    for (int l = 0; l < NB; ++l) att[l * 64 + j] *= inv;
  }
  __syncthreads();

  // msg[j,d] = sum_l att[l,j] * v1[l,d] * v2[b,l,j,d]
  const int d0 = lane * 4;
  for (int j = wave; j < NB; j += 8) {
    float a0 = 0.f, a1 = 0.f, a2 = 0.f, a3 = 0.f;
    size_t v2base = ((size_t)(b * NB) * NB + j) * QKVW + 384 + d0;
    for (int l = 0; l < NB; ++l) {
      float w = att[l * 64 + j];
      v4h x = *(const v4h*)(v1S + l * DD + d0);
      v4h z = *(const v4h*)(qkv + v2base + (size_t)l * (NB * QKVW));
      a0 += w * (float)x[0] * (float)z[0];
      a1 += w * (float)x[1] * (float)z[1];
      a2 += w * (float)x[2] * (float)z[2];
      a3 += w * (float)x[3] * (float)z[3];
    }
    v4h o;
    o[0] = (_Float16)a0; o[1] = (_Float16)a1;
    o[2] = (_Float16)a2; o[3] = (_Float16)a3;
    *(v4h*)(msg + ((size_t)(bi * NB + j)) * DD + d0) = o;
  }
}

// ---------------------------------------------------------------------------
// LayerNorm: one wave per 128-wide row (wave32, 4 elems/lane, shfl_xor tree).
// In-place on pf (f32 master), also writes the f16 GEMM-input copy.
// ---------------------------------------------------------------------------
__global__ void __launch_bounds__(256)
ln_kernel(float* __restrict__ pf, _Float16* __restrict__ pfh,
          const float* __restrict__ g, const float* __restrict__ bta) {
  const int wave = threadIdx.x >> 5;
  const int lane = threadIdx.x & 31;
  size_t row = (size_t)blockIdx.x * 8 + wave;
  float* p = pf + row * DD + lane * 4;
  float x0 = p[0], x1 = p[1], x2 = p[2], x3 = p[3];
  float s  = x0 + x1 + x2 + x3;
  float sq = x0 * x0 + x1 * x1 + x2 * x2 + x3 * x3;
  for (int off = 16; off; off >>= 1) {
    s  += __shfl_xor(s,  off);
    sq += __shfl_xor(sq, off);
  }
  float mu  = s * (1.0f / 128.0f);
  float var = sq * (1.0f / 128.0f) - mu * mu;
  float rs  = rsqrtf(var + 1e-5f);
  int c = lane * 4;
  float y0 = (x0 - mu) * rs * g[c + 0] + bta[c + 0];
  float y1 = (x1 - mu) * rs * g[c + 1] + bta[c + 1];
  float y2 = (x2 - mu) * rs * g[c + 2] + bta[c + 2];
  float y3 = (x3 - mu) * rs * g[c + 3] + bta[c + 3];
  p[0] = y0; p[1] = y1; p[2] = y2; p[3] = y3;
  v4h h;
  h[0] = (_Float16)y0; h[1] = (_Float16)y1;
  h[2] = (_Float16)y2; h[3] = (_Float16)y3;
  *(v4h*)(pfh + row * DD + c) = h;
}

// ---------------------------------------------------------------------------
// Small helpers: zero / convert / embed / scatter / diag / decode / loss
// ---------------------------------------------------------------------------
__global__ void zero_kernel(float* p, size_t n) {
  for (size_t i = (size_t)blockIdx.x * blockDim.x + threadIdx.x; i < n;
       i += (size_t)gridDim.x * blockDim.x)
    p[i] = 0.f;
}

__global__ void f32_to_f16_kernel(const float* __restrict__ s,
                                  _Float16* __restrict__ d, size_t n) {
  for (size_t i = (size_t)blockIdx.x * blockDim.x + threadIdx.x; i < n;
       i += (size_t)gridDim.x * blockDim.x)
    d[i] = (_Float16)s[i];
}

__global__ void node_embed_kernel(const float* __restrict__ x,
                                  const float* __restrict__ Wn,
                                  const float* __restrict__ bn,
                                  float* __restrict__ nf) {
  int idx = blockIdx.x * 256 + threadIdx.x;        // (B*N) * D
  if (idx >= BB * NB * DD) return;
  int r = idx >> 7, c = idx & 127;
  float acc = bn[c];
#pragma unroll
  for (int t = 0; t < 16; ++t) acc += x[r * 16 + t] * Wn[t * DD + c];
  nf[idx] = acc;
}

__global__ void edge_scatter_kernel(const float* __restrict__ ea,
                                    const float* __restrict__ We,
                                    const float* __restrict__ be,
                                    const int* __restrict__ src,
                                    const int* __restrict__ dst,
                                    float* __restrict__ pf) {
  int idx = blockIdx.x * 256 + threadIdx.x;        // E * D
  if (idx >= 32000 * DD) return;
  int e = idx >> 7, c = idx & 127;
  float acc = be[c];
#pragma unroll
  for (int t = 0; t < 4; ++t) acc += ea[e * 4 + t] * We[t * DD + c];
  int s = src[e];
  int b = s / NB, i = s % NB;
  int j = dst[e];
  atomicAdd(pf + (((size_t)(b * NB + i)) * NB + j) * DD + c, acc);
}

__global__ void diag_add_kernel(const float* __restrict__ nf,
                                float* __restrict__ pf) {
  int idx = blockIdx.x * 256 + threadIdx.x;        // (B*N) * D
  if (idx >= BB * NB * DD) return;
  int bi = idx >> 7, c = idx & 127;
  pf[((size_t)bi * NB + (bi % NB)) * DD + c] += nf[idx];
}

__global__ void __launch_bounds__(128)
decode_kernel(const float* __restrict__ pf, const float* __restrict__ Wdec,
              const float* __restrict__ bdec, const int* __restrict__ schedule,
              const float* __restrict__ y, float* __restrict__ lossvec) {
  __shared__ float red[128];
  const int b = blockIdx.x;
  const int tid = threadIdx.x;
  float accC = 0.f;
  for (int c = 0; c < 20; ++c) {
    int i = schedule[(b * 20 + c) * 2 + 0];
    int j = schedule[(b * 20 + c) * 2 + 1];
    size_t row = ((size_t)(b * NB + i)) * NB + j;
    red[tid] = pf[row * DD + tid] * Wdec[tid];
    __syncthreads();
    for (int s = 64; s > 0; s >>= 1) {
      if (tid < s) red[tid] += red[tid + s];
      __syncthreads();
    }
    if (tid == 0) accC += red[0] + bdec[0];
    __syncthreads();
  }
  if (tid == 0) lossvec[b] = fabsf(accC * (1.0f / 20.0f) - y[b]);
}

__global__ void __launch_bounds__(64)
loss_kernel(const float* __restrict__ lossvec, float* __restrict__ out) {
  __shared__ float r[64];
  int t = threadIdx.x;
  r[t] = lossvec[t];
  __syncthreads();
  for (int s = 32; s > 0; s >>= 1) {
    if (t < s) r[t] += r[t + s];
    __syncthreads();
  }
  if (t == 0) out[0] = r[0] * (1.0f / 64.0f);
}

// ---------------------------------------------------------------------------
// Host launcher
// ---------------------------------------------------------------------------
extern "C" void kernel_launch(void* const* d_in, const int* in_sizes, int n_in,
                              void* d_out, int out_size, void* d_ws, size_t ws_size,
                              hipStream_t stream) {
  (void)in_sizes; (void)n_in; (void)out_size; (void)ws_size;

  const float* node_x    = (const float*)d_in[0];
  const float* edge_attr = (const float*)d_in[1];
  const int*   edge_src  = (const int*)d_in[2];
  const int*   edge_dst  = (const int*)d_in[3];
  const int*   schedule  = (const int*)d_in[4];
  const float* yv        = (const float*)d_in[5];
  const float* Wn   = (const float*)d_in[6];
  const float* bn   = (const float*)d_in[7];
  const float* We   = (const float*)d_in[8];
  const float* be   = (const float*)d_in[9];
  const float* Wp   = (const float*)d_in[10];
  const float* bp   = (const float*)d_in[11];
  const float* Wdec = (const float*)d_in[12];
  const float* bdec = (const float*)d_in[13];

  // ---- workspace bump allocator (256B aligned) ----
  char* base = (char*)d_ws;
  size_t off = 0;
  auto alloc = [&](size_t bytes) -> void* {
    void* p = base + off;
    off = (off + bytes + 255) & ~(size_t)255;
    return p;
  };

  float*     pf   = (float*)    alloc((size_t)ROWS * DD * 4);
  _Float16*  pfh  = (_Float16*) alloc((size_t)ROWS * DD * 2);
  _Float16*  qkv  = (_Float16*) alloc((size_t)ROWS * QKVW * 2);  // also FFN hidden
  _Float16*  msgb = (_Float16*) alloc((size_t)ROWS * DD * 2);
  float*     nf   = (float*)    alloc((size_t)BB * NB * DD * 4);
  float*     lossvec = (float*) alloc(64 * 4);
  _Float16*  hbuf = qkv;        // [ROWS, 512] f16, reused for FFN hidden

  _Float16* Wp_p = (_Float16*)alloc(128 * 128 * 2);
  struct LayerPacked { _Float16 *qkv, *o, *w1, *w2; } LP[4];
  for (int l = 0; l < 4; ++l) {
    LP[l].qkv = (_Float16*)alloc(128 * 512 * 2);
    LP[l].o   = (_Float16*)alloc(128 * 128 * 2);
    LP[l].w1  = (_Float16*)alloc(128 * 512 * 2);
    LP[l].w2  = (_Float16*)alloc(512 * 128 * 2);
  }

  auto pack = [&](const float* W, _Float16* P, int K, int N) {
    int total = K * N;
    pack_w_kernel<<<(total + 255) / 256, 256, 0, stream>>>(W, P, K, N);
  };
  auto gemm = [&](const _Float16* A, const _Float16* Bp, const float* bias,
                  const float* resid, float* outF, _Float16* outH,
                  int K, int N, int relu) {
    dim3 grid(ROWS / 128, N / 128);
    gemm_wmma_kernel<<<grid, 256, 0, stream>>>(A, Bp, bias, resid, outF, outH,
                                               K, N, relu);
  };

  // ---- pack all GEMM weights (f32 -> fragment-native f16) ----
  pack(Wp, Wp_p, 128, 128);
  for (int l = 0; l < 4; ++l) {
    const int o = 14 + 13 * l;
    pack_qkv_kernel<<<(128 * 512 + 255) / 256, 256, 0, stream>>>(
        (const float*)d_in[o + 0], (const float*)d_in[o + 1],
        (const float*)d_in[o + 2], (const float*)d_in[o + 3], LP[l].qkv);
    pack((const float*)d_in[o + 4], LP[l].o,  128, 128);
    pack((const float*)d_in[o + 7], LP[l].w1, 128, 512);
    pack((const float*)d_in[o + 9], LP[l].w2, 512, 128);
  }

  const size_t PFN = (size_t)ROWS * DD;

  // ---- embeddings + PathInit scatter ----
  zero_kernel<<<4096, 256, 0, stream>>>(pf, PFN);
  node_embed_kernel<<<(BB * NB * DD + 255) / 256, 256, 0, stream>>>(node_x, Wn, bn, nf);
  edge_scatter_kernel<<<(32000 * DD + 255) / 256, 256, 0, stream>>>(
      edge_attr, We, be, edge_src, edge_dst, pf);
  diag_add_kernel<<<(BB * NB * DD + 255) / 256, 256, 0, stream>>>(nf, pf);
  f32_to_f16_kernel<<<4096, 256, 0, stream>>>(pf, pfh, PFN);

  // pf = pf0h @ Wp + bp  (write f32 master, then refresh f16 copy)
  gemm(pfh, Wp_p, bp, nullptr, pf, nullptr, 128, 128, 0);
  f32_to_f16_kernel<<<4096, 256, 0, stream>>>(pf, pfh, PFN);

  // ---- transformer layers ----
  for (int l = 0; l < 4; ++l) {
    const int o = 14 + 13 * l;
    const float* ln1_g = (const float*)d_in[o + 5];
    const float* ln1_b = (const float*)d_in[o + 6];
    const float* b1    = (const float*)d_in[o + 8];
    const float* b2    = (const float*)d_in[o + 10];
    const float* ln2_g = (const float*)d_in[o + 11];
    const float* ln2_b = (const float*)d_in[o + 12];

    // fused q|k|v1|v2 projection: A read once for all four
    gemm(pfh, LP[l].qkv, nullptr, nullptr, nullptr, qkv, 128, 512, 0);

    attn_kernel<<<BB * NB, 256, 0, stream>>>(qkv, msgb);

    // pf += msg @ Wo   (in-place residual), then LayerNorm -> pf, pfh
    gemm(msgb, LP[l].o, nullptr, pf, pf, nullptr, 128, 128, 0);
    ln_kernel<<<ROWS / 8, 256, 0, stream>>>(pf, pfh, ln1_g, ln1_b);

    // FFN: hidden = relu(pfh @ W1 + b1); pf += hidden @ W2 + b2; LN
    gemm(pfh, LP[l].w1, b1, nullptr, nullptr, hbuf, 128, 512, 1);
    gemm(hbuf, LP[l].w2, b2, pf, pf, nullptr, 512, 128, 0);
    ln_kernel<<<ROWS / 8, 256, 0, stream>>>(pf, pfh, ln2_g, ln2_b);
  }

  // ---- decode + ragged gather + L1 loss ----
  decode_kernel<<<BB, 128, 0, stream>>>(pf, Wdec, bdec, schedule, yv, lossvec);
  loss_kernel<<<1, 64, 0, stream>>>(lossvec, (float*)d_out);
}
